// SimpleMoE_56599079026713
// MI455X (gfx1250) — matrix operand order repaired and checked by reading, not compiled, not run on hardware
//
#include <hip/hip_runtime.h>
#include <stdint.h>

#define B_TOK 16384
#define D_IN  784
#define D_PAD 800
#define H_DIM 64
#define O_DIM 10
#define NEXP  16

typedef __attribute__((ext_vector_type(16))) __bf16 v16bf;
typedef __attribute__((ext_vector_type(8)))  float  v8f;

union FragAB { v16bf v; uint4 q[2]; };

__device__ __forceinline__ unsigned short f2bf(float f) {
    union { float f; unsigned int u; } c; c.f = f;
    unsigned int u = c.u;
    u += 0x7FFFu + ((u >> 16) & 1u);          // round-to-nearest-even
    return (unsigned short)(u >> 16);
}

// ---------------- W1 (NE,D,H) fp32 -> W1T (NE,H,D_PAD) bf16, zero-pad K ----
__global__ void prep_w1(const float* __restrict__ W1, unsigned short* __restrict__ W1T) {
    int idx = blockIdx.x * blockDim.x + threadIdx.x;
    if (idx >= NEXP * H_DIM * D_PAD) return;
    int k = idx % D_PAD;
    int n = (idx / D_PAD) % H_DIM;
    int e = idx / (D_PAD * H_DIM);
    float v = (k < D_IN) ? W1[((size_t)e * D_IN + k) * H_DIM + n] : 0.f;
    W1T[idx] = f2bf(v);
}

// ---------------- W2 (NE,H,O) fp32 -> W2T (NE,16,H) bf16, zero-pad O -------
__global__ void prep_w2(const float* __restrict__ W2, unsigned short* __restrict__ W2T) {
    int idx = blockIdx.x * blockDim.x + threadIdx.x;
    if (idx >= NEXP * 16 * H_DIM) return;
    int h = idx % H_DIM;
    int o = (idx / H_DIM) % 16;
    int e = idx / (16 * H_DIM);
    float v = (o < O_DIM) ? W2[((size_t)e * H_DIM + h) * O_DIM + o] : 0.f;
    W2T[idx] = f2bf(v);
}

// ---------------- gating: g = x@Wg + bg ; top-2 softmax -> dense w_full ----
__global__ void gate_kernel(const float* __restrict__ x, const float* __restrict__ Wg,
                            const float* __restrict__ bg, float* __restrict__ wfull) {
    const int wid  = threadIdx.x >> 5;
    const int lane = threadIdx.x & 31;
    const int token = blockIdx.x * (blockDim.x >> 5) + wid;
    if (token >= B_TOK) return;                         // uniform per wave

    float acc[NEXP];
#pragma unroll
    for (int e = 0; e < NEXP; ++e) acc[e] = 0.f;

    for (int d = lane; d < D_IN; d += 32) {             // coalesced x reads
        float xv = x[(size_t)token * D_IN + d];
        const float4* wr = (const float4*)(Wg + (size_t)d * NEXP);
        float4 w0 = wr[0], w1 = wr[1], w2 = wr[2], w3 = wr[3];
        acc[0]  += xv * w0.x; acc[1]  += xv * w0.y; acc[2]  += xv * w0.z; acc[3]  += xv * w0.w;
        acc[4]  += xv * w1.x; acc[5]  += xv * w1.y; acc[6]  += xv * w1.z; acc[7]  += xv * w1.w;
        acc[8]  += xv * w2.x; acc[9]  += xv * w2.y; acc[10] += xv * w2.z; acc[11] += xv * w2.w;
        acc[12] += xv * w3.x; acc[13] += xv * w3.y; acc[14] += xv * w3.z; acc[15] += xv * w3.w;
    }
#pragma unroll
    for (int e = 0; e < NEXP; ++e) {
#pragma unroll
        for (int off = 16; off > 0; off >>= 1)
            acc[e] += __shfl_xor(acc[e], off, 32);      // wave32 butterfly
    }
    if (lane == 0) {
        float g[NEXP];
#pragma unroll
        for (int e = 0; e < NEXP; ++e) g[e] = acc[e] + bg[e];
        float v1 = -3.4e38f; int i1 = 0;
#pragma unroll
        for (int e = 0; e < NEXP; ++e) if (g[e] > v1) { v1 = g[e]; i1 = e; }
        float v2 = -3.4e38f; int i2 = 0;
#pragma unroll
        for (int e = 0; e < NEXP; ++e) if (e != i1 && g[e] > v2) { v2 = g[e]; i2 = e; }
        float e2 = __expf(v2 - v1);
        float inv = 1.f / (1.f + e2);
        float* row = wfull + (size_t)token * NEXP;
#pragma unroll
        for (int e = 0; e < NEXP; ++e) row[e] = 0.f;
        row[i1] = inv;
        row[i2] = e2 * inv;
    }
}

// ---------------- fused MoE: per 16-token tile, all 16 experts -------------
__global__ void __launch_bounds__(128)
moe_kernel(const float* __restrict__ x,
           const float* __restrict__ b1,
           const float* __restrict__ b2,
           const unsigned short* __restrict__ W1T,
           const unsigned short* __restrict__ W2T,
           const float* __restrict__ wfull,
           float* __restrict__ out) {
    __shared__ __align__(16) unsigned short xs[16][D_PAD];   // x tile, bf16, K-padded
    __shared__ __align__(16) unsigned short hs[4][16][H_DIM];// per-wave hidden tile
    __shared__ float outacc[16][16];
    __shared__ float wtile[16][16];

    const int tid  = threadIdx.x;
    const int lane = tid & 31;
    const int wid  = tid >> 5;
    const int row0 = blockIdx.x * 16;

    // stage x tile fp32 -> bf16 LDS (coalesced), zero-pad K to 800
    for (int i = tid; i < 16 * D_PAD; i += 128) {
        int r = i / D_PAD, c = i - r * D_PAD;
        float v = (c < D_IN) ? x[(size_t)(row0 + r) * D_IN + c] : 0.f;
        xs[r][c] = f2bf(v);
    }
    for (int i = tid; i < 256; i += 128) {
        wtile[i >> 4][i & 15] = wfull[(size_t)(row0 + (i >> 4)) * NEXP + (i & 15)];
        outacc[i >> 4][i & 15] = 0.f;
    }
    __syncthreads();

    const int mrow    = lane & 15;
    const int hiHalf  = lane >> 4;           // 0: lanes 0-15, 1: lanes 16-31
    const int offA    = hiHalf ? 8  : 0;     // A-fragment K sub-offset
    const int offB    = hiHalf ? 8  : 0;     // B-fragment K sub-offset

    float oacc[8];
#pragma unroll
    for (int r = 0; r < 8; ++r) oacc[r] = 0.f;

    for (int ei = 0; ei < 4; ++ei) {         // 4 experts per wave, uniform
        const int e = (wid << 2) + ei;

        // ===== GEMM1: h[16,64] = relu(x_tile @ W1[e] + b1[e]) ==============
        v8f acc[4];
#pragma unroll
        for (int n = 0; n < 4; ++n) {
            float bv = b1[e * H_DIM + n * 16 + mrow];    // bias is per-column
#pragma unroll
            for (int r = 0; r < 8; ++r) acc[n][r] = bv;
        }
        for (int kk = 0; kk < D_PAD; kk += 32) {
            FragAB a;
            const unsigned short* ap = &xs[mrow][kk + offA];
            a.q[0] = *(const uint4*)(ap);
            a.q[1] = *(const uint4*)(ap + 16);
#pragma unroll
            for (int n = 0; n < 4; ++n) {
                const unsigned short* bp =
                    W1T + (size_t)(e * H_DIM + n * 16 + mrow) * D_PAD + kk + offB;
                if (kk + 32 < D_PAD)
                    __builtin_prefetch((const void*)(bp + 32), 0, 1);
                FragAB bf;
                bf.q[0] = *(const uint4*)(bp);
                bf.q[1] = *(const uint4*)(bp + 16);
                acc[n] = __builtin_amdgcn_wmma_f32_16x16x32_bf16(
                    false, a.v, false, bf.v, (short)0, acc[n], false, false);
            }
        }

        // relu -> bf16 -> LDS (re-layout D-frag -> A-frag for GEMM2)
#pragma unroll
        for (int n = 0; n < 4; ++n) {
#pragma unroll
            for (int r = 0; r < 8; ++r) {
                float v = acc[n][r];
                v = v > 0.f ? v : 0.f;
                hs[wid][r + (hiHalf << 3)][n * 16 + mrow] = f2bf(v);
            }
        }
        asm volatile("s_wait_dscnt 0" ::: "memory");     // LDS RAW within wave

        // ===== GEMM2: y[16,16] = h @ W2T[e] + b2[e] (O padded to 16) =======
        v8f yacc;
        {
            float bv = (mrow < O_DIM) ? b2[e * O_DIM + mrow] : 0.f;
#pragma unroll
            for (int r = 0; r < 8; ++r) yacc[r] = bv;
        }
#pragma unroll
        for (int kk = 0; kk < H_DIM; kk += 32) {
            FragAB a2;
            const unsigned short* ap = &hs[wid][mrow][kk + offA];
            a2.q[0] = *(const uint4*)(ap);
            a2.q[1] = *(const uint4*)(ap + 16);
            const unsigned short* bp =
                W2T + (size_t)(e * 16 + mrow) * H_DIM + kk + offB;
            FragAB bf;
            bf.q[0] = *(const uint4*)(bp);
            bf.q[1] = *(const uint4*)(bp + 16);
            yacc = __builtin_amdgcn_wmma_f32_16x16x32_bf16(
                false, a2.v, false, bf.v, (short)0, yacc, false, false);
        }

        // weighted combine: out[b,:] += w_full[b,e] * y[b,:]
#pragma unroll
        for (int r = 0; r < 8; ++r) {
            float w = wtile[r + (hiHalf << 3)][e];
            oacc[r] += w * yacc[r];
        }
    }

    // reduce partial outputs across the 4 waves of this block
#pragma unroll
    for (int r = 0; r < 8; ++r)
        atomicAdd(&outacc[r + (hiHalf << 3)][mrow], oacc[r]);   // ds_add_f32
    __syncthreads();

    for (int i = tid; i < 16 * O_DIM; i += 128) {
        int r = i / O_DIM, c = i - r * O_DIM;
        out[(size_t)(row0 + r) * O_DIM + c] = outacc[r][c];
    }
}

// ---------------------------------------------------------------------------
extern "C" void kernel_launch(void* const* d_in, const int* in_sizes, int n_in,
                              void* d_out, int out_size, void* d_ws, size_t ws_size,
                              hipStream_t stream) {
    const float* x  = (const float*)d_in[0];
    const float* Wg = (const float*)d_in[1];
    const float* bg = (const float*)d_in[2];
    const float* W1 = (const float*)d_in[3];
    const float* b1 = (const float*)d_in[4];
    const float* W2 = (const float*)d_in[5];
    const float* b2 = (const float*)d_in[6];
    float* out = (float*)d_out;

    // workspace layout (16B aligned slices)
    const size_t w1t_bytes = (size_t)NEXP * H_DIM * D_PAD * 2;   // 1,638,400
    const size_t w2t_bytes = (size_t)NEXP * 16 * H_DIM * 2;      //    32,768
    unsigned short* W1T  = (unsigned short*)d_ws;
    unsigned short* W2T  = (unsigned short*)((char*)d_ws + w1t_bytes);
    float*          wful = (float*)((char*)d_ws + w1t_bytes + w2t_bytes);

    prep_w1<<<(NEXP * H_DIM * D_PAD + 255) / 256, 256, 0, stream>>>(W1, W1T);
    prep_w2<<<(NEXP * 16 * H_DIM + 255) / 256, 256, 0, stream>>>(W2, W2T);
    gate_kernel<<<B_TOK / 8, 256, 0, stream>>>(x, Wg, bg, wful);
    moe_kernel<<<B_TOK / 16, 128, 0, stream>>>(x, b1, b2, W1T, W2T, wful, out);
}
